// Attention_16630113370932
// MI455X (gfx1250) — matrix operand order repaired and checked by reading, not compiled, not run on hardware
//
#include <hip/hip_runtime.h>

// CDNA5 / gfx1250, wave32. bf16 WMMA (v_wmma_f32_16x16x32_bf16) throughout.
// All GEMM fragments are contiguous vectorized bf16 loads (no guards, no cvt
// in hot loops): data is pre-packed into fragment-friendly layouts.

typedef __attribute__((ext_vector_type(16))) __bf16 v16bf;
typedef __attribute__((ext_vector_type(8)))  __bf16 v8bf;
typedef __attribute__((ext_vector_type(8)))  float  v8f;

#define NH    12
#define DH    64
#define NTOK  196
#define NPAD  224     // 14*16, multiple of 32 -> clean WMMA K loops
#define CIN   768
#define OQKV  2304
#define BATCH 64

union U16 { v16bf v; v8bf h[2]; };

__device__ __forceinline__ __bf16 tobf(float f) { return (__bf16)f; }

static __device__ __forceinline__ v8f wmma_bf16(v16bf a, v16bf b, v8f c) {
    return __builtin_amdgcn_wmma_f32_16x16x32_bf16(false, a, false, b, (short)0, c,
                                                   false, false);
}

// ---------------------------------------------------------------------------
// Pack: weights fp32 -> bf16 (coalesced elementwise)
// ---------------------------------------------------------------------------
__global__ __launch_bounds__(256)
void pack_w(const float* __restrict__ src, __bf16* __restrict__ dst, int nelem) {
    const int i = blockIdx.x * 256 + threadIdx.x;
    if (i < nelem) dst[i] = tobf(src[i]);
}

// Pack x [B,768,196] fp32 -> Xt [B,224,768] bf16, token-major, zero padding.
__global__ __launch_bounds__(256)
void pack_x(const float* __restrict__ x, __bf16* __restrict__ Xt) {
    int i = blockIdx.x * 256 + threadIdx.x;       // B*768*224, tok fastest (coalesced reads)
    const int tok = i % NPAD; i /= NPAD;
    const int c   = i % CIN;
    const int b   = i / CIN;
    const float v = (tok < NTOK) ? x[((size_t)b * CIN + c) * NTOK + tok] : 0.0f;
    Xt[((size_t)b * NPAD + tok) * CIN + c] = tobf(v);
}

// ---------------------------------------------------------------------------
// Stage 1: qkv = Wq @ x   (per b: [2304x768] x [768x224])
// A-frag: two v8bf from Wq row.  B-frag: one v16bf from Xt (token-major).
// Writes Qb/Kb [bh][tok][d] (v8bf stores) and Vt [bh][d][tok].
// ---------------------------------------------------------------------------
__global__ __launch_bounds__(256)
void qkv_kernel(const __bf16* __restrict__ Xt, const __bf16* __restrict__ Wq,
                __bf16* __restrict__ Qb, __bf16* __restrict__ Kb,
                __bf16* __restrict__ Vt) {
    const int lane = threadIdx.x;
    int g = blockIdx.x * 8 + threadIdx.y;          // one 16x16 tile per wave
    const int nt = g % 14;  g /= 14;
    const int ot = g % 144; g /= 144;
    const int b  = g;
    const int l15 = lane & 15;
    const int hiA = (lane & 16) ? 16 : 0;
    const int hiB = (lane & 16) ? 16 : 0;
    const int n = nt * 16 + l15;
    const __bf16* arow = Wq + (size_t)(ot * 16 + l15) * CIN + hiA;
    const __bf16* brow = Xt + ((size_t)b * NPAD + n) * CIN + hiB;

    v8f c = {};
    for (int kk = 0; kk < CIN; kk += 32) {
        U16 a;
        a.h[0] = *(const v8bf*)(arow + kk);
        a.h[1] = *(const v8bf*)(arow + kk + 8);
        const v16bf bb = *(const v16bf*)(brow + kk);
        c = wmma_bf16(a.v, bb, c);
    }
    // Rows o0..o0+7 -> 8 consecutive d within one head; same q/k/v selector.
    const int mhi = (lane & 16) ? 8 : 0;
    const int o0 = ot * 16 + mhi;
    const int wq = o0 / CIN;
    const int oc = o0 % CIN;
    const int h  = oc >> 6;
    const int d0 = oc & 63;                        // multiple of 8
    const size_t bh = (size_t)b * NH + h;
    union { v8bf v; __bf16 e[8]; } r;
    #pragma unroll
    for (int j = 0; j < 8; ++j) r.e[j] = tobf(c[j]);
    if (wq == 0) {
        *(v8bf*)(Qb + (bh * NPAD + n) * DH + d0) = r.v;
    } else if (wq == 1) {
        *(v8bf*)(Kb + (bh * NPAD + n) * DH + d0) = r.v;
    } else {
        #pragma unroll
        for (int j = 0; j < 8; ++j)
            Vt[(bh * DH + d0 + j) * NPAD + n] = r.e[j];
    }
}

// ---------------------------------------------------------------------------
// Stage 2: per (b,h): logits = 0.125 * Q K^T, in-register row softmax,
// P @ V.  One workgroup per (b,h); each wave owns 16 query rows and the
// full 224-key stripe in 14 f32 C-fragments.
// ---------------------------------------------------------------------------
__global__ __launch_bounds__(256)
void attn_kernel(const __bf16* __restrict__ Qb, const __bf16* __restrict__ Kb,
                 const __bf16* __restrict__ Vt, __bf16* __restrict__ AOt) {
    __shared__ __bf16 plds[8][16][NPAD];           // per-wave P row-block, bf16
    const int lane = threadIdx.x;
    const int w    = threadIdx.y;
    const int bh   = blockIdx.x;
    const int b = bh / NH, h = bh % NH;
    const int l15 = lane & 15;
    const int hiA = (lane & 16) ? 16 : 0;
    const int hiB = (lane & 16) ? 16 : 0;
    const int mhi = (lane & 16) ? 8 : 0;
    const __bf16* Qbase = Qb + (size_t)bh * NPAD * DH;
    const __bf16* Kbase = Kb + (size_t)bh * NPAD * DH;
    const __bf16* Vbase = Vt + (size_t)bh * DH * NPAD;

    for (int rb = w; rb < 13; rb += 8) {           // 13 row-blocks cover 196 queries
        const int q0 = rb * 16;
        const __bf16* qr = Qbase + (size_t)(q0 + l15) * DH + hiA;
        U16 aq0, aq1;                              // K-dim = dh = 64 -> two steps
        aq0.h[0] = *(const v8bf*)(qr);
        aq0.h[1] = *(const v8bf*)(qr + 8);
        aq1.h[0] = *(const v8bf*)(qr + 32);
        aq1.h[1] = *(const v8bf*)(qr + 40);

        v8f acc[14];
        #pragma unroll
        for (int mt = 0; mt < 14; ++mt) {
            const int key = mt * 16 + l15;
            const __bf16* kr = Kbase + (size_t)key * DH + hiB;
            const v16bf bk0 = *(const v16bf*)(kr);        // d = hiB+0..15
            const v16bf bk1 = *(const v16bf*)(kr + 32);   // d = 32+hiB+0..15
            v8f t = {};
            t = wmma_bf16(aq0.v, bk0, t);
            t = wmma_bf16(aq1.v, bk1, t);
            acc[mt] = t;
        }
        // Row softmax: row M=j+mhi lives in VGPR j across one 16-lane half.
        #pragma unroll
        for (int j = 0; j < 8; ++j) {
            float mx = -1e30f;
            #pragma unroll
            for (int mt = 0; mt < 14; ++mt) {
                const int key = mt * 16 + l15;
                const float lg = (key < NTOK) ? acc[mt][j] * 0.125f : -1e30f;
                acc[mt][j] = lg;
                mx = fmaxf(mx, lg);
            }
            #pragma unroll
            for (int off = 1; off < 16; off <<= 1)
                mx = fmaxf(mx, __shfl_xor(mx, off, 16));
            float sm = 0.0f;
            #pragma unroll
            for (int mt = 0; mt < 14; ++mt) {
                const float e = __expf(acc[mt][j] - mx);  // masked -> exp(-1e30)=0
                acc[mt][j] = e;
                sm += e;
            }
            #pragma unroll
            for (int off = 1; off < 16; off <<= 1)
                sm += __shfl_xor(sm, off, 16);
            const float inv = 1.0f / sm;
            #pragma unroll
            for (int mt = 0; mt < 14; ++mt) acc[mt][j] *= inv;
        }
        // C-layout -> A-layout via LDS (bf16). Wave-local: DS ops in-order.
        #pragma unroll
        for (int mt = 0; mt < 14; ++mt) {
            const int key = mt * 16 + l15;
            #pragma unroll
            for (int j = 0; j < 8; ++j)
                plds[w][j + mhi][key] = tobf(acc[mt][j]);
        }
        // out(16x64) = P(16x224) @ V(224x64); V B-frags contiguous in Vt.
        const __bf16* pr = &plds[w][l15][hiA];
        #pragma unroll
        for (int dt = 0; dt < 4; ++dt) {
            const int d = dt * 16 + l15;
            const __bf16* vr = Vbase + (size_t)d * NPAD + hiB;
            v8f o2 = {};
            #pragma unroll
            for (int kt = 0; kt < 7; ++kt) {
                U16 ap;
                ap.h[0] = *(const v8bf*)(pr + kt * 32);        // ds_load_b128
                ap.h[1] = *(const v8bf*)(pr + kt * 32 + 8);
                const v16bf bv = *(const v16bf*)(vr + kt * 32);
                o2 = wmma_bf16(ap.v, bv, o2);
            }
            const int ch = h * 64 + d;
            #pragma unroll
            for (int j = 0; j < 8; ++j) {
                const int tok = q0 + j + mhi;
                AOt[((size_t)b * NPAD + tok) * CIN + ch] = tobf(o2[j]);
            }
        }
    }
}

// ---------------------------------------------------------------------------
// Stage 3: out = Wp @ AO  (per b: [768x768] x [768x224]), fp32 output.
// AOt token-major -> unconditional v16bf B-frags (padding is finite/zero).
// ---------------------------------------------------------------------------
__global__ __launch_bounds__(256)
void proj_kernel(const __bf16* __restrict__ Wp, const __bf16* __restrict__ AOt,
                 float* __restrict__ out) {
    const int lane = threadIdx.x;
    int g = blockIdx.x * 8 + threadIdx.y;
    const int nt = g % 13; g /= 13;
    const int ot = g % 48; g /= 48;
    const int b  = g;
    const int l15 = lane & 15;
    const int hiA = (lane & 16) ? 16 : 0;
    const int hiB = (lane & 16) ? 16 : 0;
    const int n = nt * 16 + l15;
    const __bf16* arow = Wp  + (size_t)(ot * 16 + l15) * CIN + hiA;
    const __bf16* brow = AOt + ((size_t)b * NPAD + n) * CIN + hiB;

    v8f c = {};
    for (int kk = 0; kk < CIN; kk += 32) {
        U16 a;
        a.h[0] = *(const v8bf*)(arow + kk);
        a.h[1] = *(const v8bf*)(arow + kk + 8);
        const v16bf bb = *(const v16bf*)(brow + kk);
        c = wmma_bf16(a.v, bb, c);
    }
    if (n < NTOK) {
        const int mhi = (lane & 16) ? 8 : 0;
        #pragma unroll
        for (int j = 0; j < 8; ++j)
            out[((size_t)b * CIN + ot * 16 + j + mhi) * NTOK + n] = c[j];
    }
}

extern "C" void kernel_launch(void* const* d_in, const int* in_sizes, int n_in,
                              void* d_out, int out_size, void* d_ws, size_t ws_size,
                              hipStream_t stream) {
    const float* x      = (const float*)d_in[0];   // [64,768,14,14]
    const float* w_qkv  = (const float*)d_in[1];   // [2304,768]
    const float* w_proj = (const float*)d_in[2];   // [768,768]
    float* out = (float*)d_out;                    // [64,768,14,14]

    const size_t SEG = (size_t)BATCH * NPAD * CIN;   // 11,010,048 bf16 elems
    __bf16* Xt = (__bf16*)d_ws;                      // [64,224,768]; reused as AOt
    __bf16* Wq = Xt + SEG;                           // [2304,768]
    __bf16* Wp = Wq + (size_t)OQKV * CIN;            // [768,768]
    __bf16* Qb = Wp + (size_t)CIN * CIN;             // [64,12,224,64]
    __bf16* Kb = Qb + SEG;                           // [64,12,224,64]
    __bf16* Vt = Kb + SEG;                           // [64,12,64,224]
    __bf16* AOt = Xt;                                // alias: Xt dead after stage 1

    dim3 blk(32, 8);                                 // 8 wave32 per block
    pack_x<<<43008, 256, 0, stream>>>(x, Xt);                        // 64*768*224
    pack_w<<< 6912, 256, 0, stream>>>(w_qkv, Wq, OQKV * CIN);
    pack_w<<< 2304, 256, 0, stream>>>(w_proj, Wp, CIN * CIN);
    qkv_kernel <<<16128, blk, 0, stream>>>(Xt, Wq, Qb, Kb, Vt);      // 64*144*14 waves
    attn_kernel<<<  768, blk, 0, stream>>>(Qb, Kb, Vt, AOt);         // one WG per (b,h)
    proj_kernel<<< 4992, blk, 0, stream>>>(Wp, AOt, out);            // 64*48*13 waves
}
